// LSTM_89678917141105
// MI455X (gfx1250) — matrix operand order repaired and checked by reading, not compiled, not run on hardware
//
#include <hip/hip_runtime.h>

// ---------------------------------------------------------------------------
// LSTM on MI455X (gfx1250): bf16 WMMA, f32 accumulate.
//   Phase A: pack weights to WMMA B-fragment layout, X -> bf16
//   Phase B: Z = X@Wi + b   (parallel WMMA GEMM, Wi tile staged to LDS via
//            global_load_async_to_lds_b128)
//   Phase C: persistent 32-WG kernel; each WG DMAs its 256KB Wh slice into
//            LDS once via the Tensor Data Mover, keeps it resident for all
//            512 steps; h double-buffered in global; cell state in LDS;
//            one grid barrier per step; h.Wd projection folded in.
// ---------------------------------------------------------------------------

typedef __bf16 v16bf __attribute__((ext_vector_type(16)));
typedef float  v8f   __attribute__((ext_vector_type(8)));
typedef unsigned short us8 __attribute__((ext_vector_type(8)));
typedef unsigned int u32x4 __attribute__((ext_vector_type(4)));
typedef int i32x8 __attribute__((ext_vector_type(8)));
typedef int i32x4 __attribute__((ext_vector_type(4)));

union Frag16 { v16bf v; us8 h[2]; };

constexpr int BB = 32;     // batch
constexpr int TT = 512;    // time
constexpr int DD = 512;    // input dim
constexpr int FF = 1024;   // hidden dim
constexpr int GG = 4096;   // 4*F gate columns
constexpr int NWG = 32;    // persistent workgroups (32 hidden units each)

__device__ __forceinline__ unsigned short f2bf(float f) {
  unsigned u = __float_as_uint(f);
  u += 0x7FFFu + ((u >> 16) & 1u);          // round-to-nearest-even
  return (unsigned short)(u >> 16);
}
__device__ __forceinline__ float bf2f(unsigned short s) {
  return __uint_as_float(((unsigned)s) << 16);
}
__device__ __forceinline__ float sigmoidf_(float x) {
  return 1.0f / (1.0f + __expf(-x));
}

// ---- CDNA5 async copy: global -> LDS, 16B per lane -------------------------
// LDS flat addressing uses addr[31:0] (ISA 10.2), so truncating the generic
// shared pointer gives the LDS byte offset the instruction needs.
__device__ __forceinline__ void asyncLds16(void* ldsPtr, const void* gPtr) {
  unsigned lds = (unsigned)(uintptr_t)ldsPtr;
  unsigned long long ga = (unsigned long long)(uintptr_t)gPtr;
  asm volatile("global_load_async_to_lds_b128 %0, %1, off"
               :: "v"(lds), "v"(ga) : "memory");
}
__device__ __forceinline__ void waitAsync0() {
  asm volatile("s_wait_asynccnt 0x0" ::: "memory");
}

// ---- CDNA5 Tensor Data Mover: 64KB contiguous global -> LDS ---------------
// D# per ISA 08_async_tensor 8.3/8.4: 1-D tensor, data_size=8B (code 3),
// tensor_dim0 = tile_dim0 = 8192 units. Groups 2/3 zero (<=2D tensor).
// 6-arg builtin form (clang-23 / therock-10.0 headers).
__device__ __forceinline__ void tdmLoad64KB(void* ldsPtr, const void* gPtr) {
  unsigned lds = (unsigned)(uintptr_t)ldsPtr;
  unsigned long long ga = (unsigned long long)(uintptr_t)gPtr;
  u32x4 g0;
  g0[0] = 1u;                                          // count=1, user D#
  g0[1] = lds;                                         // lds_addr (bytes)
  g0[2] = (unsigned)(ga & 0xFFFFFFFFu);                // global_addr[31:0]
  g0[3] = (unsigned)((ga >> 32) & 0x01FFFFFFu) | (2u << 30);  // [56:32], type=2
  i32x8 g1 = {0, 0, 0, 0, 0, 0, 0, 0};
  g1[0] = (3 << 16);                                   // data_size=8B, mask=0
  g1[1] = (int)((8192u & 0xFFFFu) << 16);              // tensor_dim0[15:0] @bits63:48
  g1[2] = (int)((8192u >> 16) & 0xFFFFu);              // tensor_dim0[31:16] @bits79:64
  g1[3] = (int)(8192u << 16);                          // tile_dim0 @bits127:112
  i32x4 zz4 = {0, 0, 0, 0};
  i32x8 zz8 = {0, 0, 0, 0, 0, 0, 0, 0};
  __builtin_amdgcn_tensor_load_to_lds(g0, g1, zz4, zz4, zz8, 0);
}

// ---------------------------------------------------------------------------
// sense-reversing grid barrier (NWG co-resident blocks), agent scope
// ---------------------------------------------------------------------------
__device__ __forceinline__ void gridBarrier(unsigned* cnt, unsigned* gen, unsigned nb) {
  __syncthreads();
  if (threadIdx.x == 0) {
    unsigned g = __hip_atomic_load(gen, __ATOMIC_RELAXED, __HIP_MEMORY_SCOPE_AGENT);
    unsigned prev = __hip_atomic_fetch_add(cnt, 1u, __ATOMIC_ACQ_REL, __HIP_MEMORY_SCOPE_AGENT);
    if (prev == nb - 1u) {
      __hip_atomic_store(cnt, 0u, __ATOMIC_RELAXED, __HIP_MEMORY_SCOPE_AGENT);
      __hip_atomic_fetch_add(gen, 1u, __ATOMIC_RELEASE, __HIP_MEMORY_SCOPE_AGENT);
    } else {
      while (__hip_atomic_load(gen, __ATOMIC_ACQUIRE, __HIP_MEMORY_SCOPE_AGENT) == g)
        __builtin_amdgcn_s_sleep(2);
    }
  }
  __syncthreads();
}

// ---------------------------------------------------------------------------
// prep kernels
// ---------------------------------------------------------------------------
__global__ void k_cvt_bf16(const float* __restrict__ src, unsigned short* __restrict__ dst, int n) {
  int i = blockIdx.x * 256 + threadIdx.x;
  if (i < n) dst[i] = f2bf(src[i]);
}

// Pack W [K, 4096] row-major fp32 into WMMA B-fragment order, bf16:
//   P[((ct*KB + kb)*32 + lane)*16 + e] = W[k, n]
//   n = ct*16 + lane%16 ;  k = kb*32 + (lane/16)*16 + e   (B: K = 16*hi + e)
__global__ void k_pack_w(const float* __restrict__ W, unsigned short* __restrict__ P,
                         int K, int total) {
  int i = blockIdx.x * 256 + threadIdx.x;
  if (i >= total) return;
  const int KB = K >> 5;
  const int e = i & 15;
  const int lane = (i >> 4) & 31;
  const int rest = i >> 9;
  const int kb = rest % KB;
  const int ct = rest / KB;
  const int n = ct * 16 + (lane & 15);
  const int k = kb * 32 + (lane >> 4) * 16 + e;
  P[i] = f2bf(W[(size_t)k * GG + n]);
}

__global__ void k_init_state(unsigned short* hb0, unsigned short* hb1, unsigned* bar) {
  int i = blockIdx.x * 256 + threadIdx.x;
  if (i < BB * FF) { hb0[i] = 0; hb1[i] = 0; }
  if (i < 2) bar[i] = 0;
}

// ---------------------------------------------------------------------------
// Phase B: Z[16384,4096] = Xbf16 @ WiP + b   (bf16 out)
// block = 256 thr = 8 waves; tile 128 rows x 64 cols; Wi slice (64KB) in LDS.
// ---------------------------------------------------------------------------
__global__ __launch_bounds__(256) void k_gemm_xwi(
    const unsigned short* __restrict__ Xb,
    const unsigned short* __restrict__ WiP,
    const float* __restrict__ bias,
    unsigned short* __restrict__ Z) {
  extern __shared__ char smem[];                  // 64KB: [4 ct][16 kb][32][16]
  unsigned short* wiL = (unsigned short*)smem;
  constexpr int KB = DD / 32;

  const int tid = threadIdx.x;
  const int wv = tid >> 5, lane = tid & 31;
  const int l = lane & 15, hi = lane >> 4;
  const int rowBase = blockIdx.x * 128 + wv * 16;
  const int cb = blockIdx.y;

  // async-stage this block's 64KB Wi slice into LDS
  {
    const char* src = (const char*)(WiP + (size_t)(cb * 4) * KB * 512);
    for (int off = tid * 16; off < 4 * KB * 1024; off += 256 * 16)
      asyncLds16(smem + off, src + off);
    waitAsync0();
  }
  __syncthreads();

  const v8f zv = {0.f, 0.f, 0.f, 0.f, 0.f, 0.f, 0.f, 0.f};
  v8f acc[4] = {zv, zv, zv, zv};

  const unsigned short* xr = Xb + (size_t)(rowBase + l) * DD;
  Frag16 af[2], bfr[2][4];
  auto loadA = [&](int buf, int kb) {
    const int k0 = kb * 32 + 16 * hi;             // A: K = 16*hi+e
    af[buf].h[0] = *(const us8*)(xr + k0);
    af[buf].h[1] = *(const us8*)(xr + k0 + 8);
  };
  auto loadB = [&](int buf, int kb) {
#pragma unroll
    for (int nt = 0; nt < 4; ++nt) {
      const unsigned short* bp = wiL + ((nt * KB + kb) * 32 + lane) * 16;
      bfr[buf][nt].h[0] = *(const us8*)bp;        // ds_load_b128
      bfr[buf][nt].h[1] = *(const us8*)(bp + 8);
    }
  };
  loadA(0, 0);
  loadB(0, 0);
  for (int kb = 0; kb < KB; ++kb) {
    const int cur = kb & 1;
    if (kb + 1 < KB) {                            // prefetch next k-step
      loadA(cur ^ 1, kb + 1);
      loadB(cur ^ 1, kb + 1);
    }
#pragma unroll
    for (int nt = 0; nt < 4; ++nt)
      acc[nt] = __builtin_amdgcn_wmma_f32_16x16x32_bf16(
          false, af[cur].v, false, bfr[cur][nt].v, (short)0, acc[nt], false, false);
  }
#pragma unroll
  for (int nt = 0; nt < 4; ++nt) {
    const int col = cb * 64 + nt * 16 + l;
    const float bv = bias[col];
#pragma unroll
    for (int r = 0; r < 8; ++r) {
      const int row = rowBase + r + 8 * hi;       // C: M = r + 8*hi, N = lane%16
      Z[(size_t)row * GG + col] = f2bf(acc[nt][r] + bv);
    }
  }
}

// ---------------------------------------------------------------------------
// Phase C: persistent recurrent kernel. 32 WGs x 256 thr (8 waves).
// WG w owns hidden units [w*32, w*32+32) across ALL FOUR gates.
// Its Wh slice (8 N-tiles x 32 k-blocks = 256KB) is TDM-loaded to LDS once.
// wave (m = wv&1, gate = wv>>1) computes 16 rows x 32 cols of one gate.
// ---------------------------------------------------------------------------
__global__ __launch_bounds__(256, 1) void k_lstm_seq(
    const unsigned short* __restrict__ Z,
    const unsigned short* __restrict__ WhP,
    const float* __restrict__ Wd,
    const float* __restrict__ bd,
    unsigned short* __restrict__ hb0,
    unsigned short* __restrict__ hb1,
    float* __restrict__ pbuf,          // [2][NWG*32] partial h.Wd
    unsigned* __restrict__ barCnt,
    unsigned* __restrict__ barGen,
    float* __restrict__ out) {
  extern __shared__ char smem[];
  unsigned short* whL = (unsigned short*)smem;             // 256KB [8 lc][32 kb][32][16]
  float* zs = (float*)(smem + 262144);                     // [4][32][32] 16KB
  float* cs = (float*)(smem + 262144 + 16384);             // [32][32]    4KB
  float* ps = (float*)(smem + 262144 + 16384 + 4096);      // [32][8]     1KB
  constexpr int KB = FF / 32;

  const int tid = threadIdx.x;
  const int wv = tid >> 5, lane = tid & 31;
  const int l = lane & 15, hi = lane >> 4;
  const int m = wv & 1, gate = wv >> 1;
  const int wg = blockIdx.x;

  // ---- TDM: stage Wh slice into LDS (4 contiguous 64KB chunks, 1 per gate)
  if (wv == 0) {
#pragma unroll
    for (int g = 0; g < 4; ++g) {
      const char* src = (const char*)(WhP + (size_t)(g * 64 + wg * 2) * KB * 512);
      tdmLoad64KB(smem + g * 65536, src);
    }
    __builtin_amdgcn_s_wait_tensorcnt(0);
  }
  __syncthreads();
  for (int i = tid; i < 32 * 32; i += 256) cs[i] = 0.0f;   // zero cell state
  __syncthreads();

  const int aRow = m * 16 + l;
  const v8f zv = {0.f, 0.f, 0.f, 0.f, 0.f, 0.f, 0.f, 0.f};

  for (int t = 0; t < TT; ++t) {
    // deterministic output write for step t-1 (parity-safe via barrier below)
    if (wg == 0 && t > 0 && tid < 32) {
      const float* pp = pbuf + ((t - 1) & 1) * (NWG * 32);
      float s = 0.0f;
#pragma unroll
      for (int k = 0; k < NWG; ++k) s += pp[k * 32 + tid];
      out[tid * TT + (t - 1)] = s + bd[0];
    }

    const unsigned short* hc = (t & 1) ? hb1 : hb0;
    unsigned short* hn = (t & 1) ? hb0 : hb1;

    // ---- GEMM slice: z += h @ Wh  (K=1024, 32 k-steps, 2 N-tiles, B in LDS)
    v8f acc[2] = {zv, zv};
    const unsigned short* hrow = hc + (size_t)aRow * FF;
    Frag16 af[2], bfr[2][2];
    auto loadA = [&](int buf, int kb) {
      const int k0 = kb * 32 + 16 * hi;
      af[buf].h[0] = *(const us8*)(hrow + k0);
      af[buf].h[1] = *(const us8*)(hrow + k0 + 8);
    };
    auto loadB = [&](int buf, int kb) {
#pragma unroll
      for (int nt = 0; nt < 2; ++nt) {
        const int lc = gate * 2 + nt;             // local N-tile in LDS
        const unsigned short* bp = whL + ((lc * KB + kb) * 32 + lane) * 16;
        bfr[buf][nt].h[0] = *(const us8*)bp;      // ds_load_b128
        bfr[buf][nt].h[1] = *(const us8*)(bp + 8);
      }
    };
    loadA(0, 0);
    loadB(0, 0);
    for (int kb = 0; kb < KB; ++kb) {
      const int cur = kb & 1;
      if (kb + 1 < KB) {                          // prefetch next k-step
        loadA(cur ^ 1, kb + 1);
        loadB(cur ^ 1, kb + 1);
      }
#pragma unroll
      for (int nt = 0; nt < 2; ++nt)
        acc[nt] = __builtin_amdgcn_wmma_f32_16x16x32_bf16(
            false, af[cur].v, false, bfr[cur][nt].v, (short)0, acc[nt], false, false);
    }

    // ---- epilogue: + Z (x@Wi + b), stage gate pre-activations into LDS
#pragma unroll
    for (int nt = 0; nt < 2; ++nt) {
      const int colL = nt * 16 + l;               // [0,32)
      const int gcol = gate * FF + wg * 32 + colL;
#pragma unroll
      for (int r = 0; r < 8; ++r) {
        const int bb = m * 16 + r + 8 * hi;
        const size_t zi = ((size_t)bb * TT + t) * GG + gcol;
        const float za = bf2f(Z[zi]);
        zs[gate * 1024 + bb * 32 + colL] = acc[nt][r] + za;
        if (t + 1 < TT && r == 0)                 // global_prefetch next step's Z
          __builtin_prefetch(&Z[zi + GG], 0, 1);
      }
    }
    __syncthreads();

    // ---- gates + cell update: thread -> (row = tid/8, 4 cols)
    {
      const int row = tid >> 3;
      const int c0 = (tid & 7) * 4;
      float pp = 0.0f;
#pragma unroll
      for (int j = 0; j < 4; ++j) {
        const int col = c0 + j;
        const float iv = sigmoidf_(zs[0 * 1024 + row * 32 + col]);
        const float fv = sigmoidf_(zs[1 * 1024 + row * 32 + col]);
        const float gv = tanhf(zs[2 * 1024 + row * 32 + col]);
        const float ov = sigmoidf_(zs[3 * 1024 + row * 32 + col]);
        const float cn = fv * cs[row * 32 + col] + iv * gv;
        cs[row * 32 + col] = cn;
        const float hv = ov * tanhf(cn);
        hn[(size_t)row * FF + wg * 32 + col] = f2bf(hv);
        pp += hv * Wd[wg * 32 + col];
      }
      ps[row * 8 + (tid & 7)] = pp;
    }
    __syncthreads();
    if (tid < 32) {
      float s = 0.0f;
#pragma unroll
      for (int j = 0; j < 8; ++j) s += ps[tid * 8 + j];
      pbuf[(t & 1) * (NWG * 32) + wg * 32 + tid] = s;
    }

    __threadfence();                              // release h / pbuf device-wide
    gridBarrier(barCnt, barGen, NWG);
    __builtin_amdgcn_fence(__ATOMIC_ACQUIRE, "agent");  // drop stale h lines
  }

  if (wg == 0 && tid < 32) {                      // flush final column t = TT-1
    const float* pp = pbuf + ((TT - 1) & 1) * (NWG * 32);
    float s = 0.0f;
#pragma unroll
    for (int k = 0; k < NWG; ++k) s += pp[k * 32 + tid];
    out[tid * TT + (TT - 1)] = s + bd[0];
  }
}

// ---------------------------------------------------------------------------
extern "C" void kernel_launch(void* const* d_in, const int* in_sizes, int n_in,
                              void* d_out, int out_size, void* d_ws, size_t ws_size,
                              hipStream_t stream) {
  const float* x  = (const float*)d_in[0];   // [B,T,D]
  const float* Wi = (const float*)d_in[1];   // [D,4F]
  const float* Wh = (const float*)d_in[2];   // [F,4F]
  const float* b  = (const float*)d_in[3];   // [4F]
  const float* Wd = (const float*)d_in[4];   // [F,1]
  const float* bd = (const float*)d_in[5];   // [1]
  float* out = (float*)d_out;                // [B,T,1]

  char* ws = (char*)d_ws;
  size_t o = 0;
  auto carve = [&](size_t bytes) -> void* {
    void* p = (void*)(ws + o);
    o += (bytes + 255) & ~(size_t)255;
    return p;
  };
  unsigned short* Xb  = (unsigned short*)carve((size_t)BB * TT * DD * 2);  // 16 MB
  unsigned short* WiP = (unsigned short*)carve((size_t)DD * GG * 2);       // 4 MB
  unsigned short* WhP = (unsigned short*)carve((size_t)FF * GG * 2);       // 8 MB
  unsigned short* Zb  = (unsigned short*)carve((size_t)BB * TT * GG * 2);  // 128 MB
  unsigned short* hb0 = (unsigned short*)carve((size_t)BB * FF * 2);
  unsigned short* hb1 = (unsigned short*)carve((size_t)BB * FF * 2);
  float*    pbuf = (float*)carve(2 * NWG * 32 * sizeof(float));
  unsigned* bar  = (unsigned*)carve(256);

  const int nX = BB * TT * DD;
  k_cvt_bf16<<<(nX + 255) / 256, 256, 0, stream>>>(x, Xb, nX);

  const int nWi = DD * GG;
  k_pack_w<<<(nWi + 255) / 256, 256, 0, stream>>>(Wi, WiP, DD, nWi);
  const int nWh = FF * GG;
  k_pack_w<<<(nWh + 255) / 256, 256, 0, stream>>>(Wh, WhP, FF, nWh);

  k_init_state<<<(BB * FF + 255) / 256, 256, 0, stream>>>(hb0, hb1, bar);

  // Phase B: Wi tile (64KB) staged to LDS per block
  k_gemm_xwi<<<dim3(BB * TT / 128, GG / 64), 256, 65536, stream>>>(Xb, WiP, b, Zb);

  // Phase C: 256KB Wh slice + 21KB working set of dynamic LDS per WG
  const size_t ldsC = 262144 + 16384 + 4096 + 1024;
  k_lstm_seq<<<NWG, 256, ldsC, stream>>>(Zb, WhP, Wd, bd, hb0, hb1, pbuf,
                                         bar, bar + 1, out);
}
